// InterestDict_71511205478458
// MI455X (gfx1250) — matrix operand order repaired and checked by reading, not compiled, not run on hardware
//
#include <hip/hip_runtime.h>

typedef _Float16 v16h __attribute__((ext_vector_type(16)));
typedef _Float16 v8h  __attribute__((ext_vector_type(8)));
typedef float    v8f  __attribute__((ext_vector_type(8)));
typedef float    v4f  __attribute__((ext_vector_type(4)));

#define D_DIM   256
#define TOPK    8
#define LDS_PAD 264            // 256 halves + 8 pad -> row stride 528B, 4-bank lane stride
#define TILE_H  (16 * LDS_PAD) // halves per tile buffer

// ---------------------------------------------------------------------------
// Prep: split embed rows into f16 hi/lo and compute squared row norms (fp32).
// ---------------------------------------------------------------------------
__global__ __launch_bounds__(256) void prep_embed(
    const float* __restrict__ embed,
    float* __restrict__ enorm,
    _Float16* __restrict__ eh,
    _Float16* __restrict__ el)
{
    __shared__ float red[256];
    const int e = blockIdx.x;
    const int d = threadIdx.x;
    float v = embed[(size_t)e * D_DIM + d];
    _Float16 hi = (_Float16)v;
    _Float16 lo = (_Float16)(v - (float)hi);
    eh[(size_t)e * D_DIM + d] = hi;
    el[(size_t)e * D_DIM + d] = lo;
    red[d] = v * v;
    __syncthreads();
    for (int s = 128; s > 0; s >>= 1) {
        if (d < s) red[d] += red[d + s];
        __syncthreads();
    }
    if (d == 0) enorm[e] = red[0];
}

// ---------------------------------------------------------------------------
// Main fused kernel: distances via split-f16 WMMA with async double-buffered
// LDS staging, per-lane register top-8, lane-pair merge, then
// group_emb = sum(e^2)/sum(e) over the selected 8 embeddings.
// Block = 256 threads = 8 waves; each wave owns 16 x-rows (the WMMA N dim).
// A operand = embed tile (M dim), B operand = x rows (N dim).
// ---------------------------------------------------------------------------
__global__ __launch_bounds__(256) void interest_main(
    const float*    __restrict__ x,       // [N,256] fp32
    const float*    __restrict__ embed,   // [M,256] fp32 (for group_emb)
    const float*    __restrict__ enorm,   // [M]
    const _Float16* __restrict__ eh,      // [M,256]
    const _Float16* __restrict__ el,      // [M,256]
    float* __restrict__ outg,             // [N,256]
    int*   __restrict__ outi,             // [N,8]
    int M)
{
    __shared__ _Float16 ehT[2 * TILE_H];
    __shared__ _Float16 elT[2 * TILE_H];
    __shared__ float md[8][32][TOPK];
    __shared__ int   mi[8][32][TOPK];
    __shared__ int   ridx[8][16][TOPK];

    const int tid  = threadIdx.x;
    const int wave = tid >> 5;
    const int lane = tid & 31;
    const int lrow = lane & 15;   // B column (x row in tile) / A row (embed in tile)
    const int half = lane >> 4;
    const int rowbase = blockIdx.x * 128 + wave * 16;
    const int xrow    = rowbase + lrow;

    // Per-thread staging assignment: 32B of hi + 32B of lo per tile.
    const int srow = tid >> 4, sseg = tid & 15;
    const size_t gstage = (size_t)srow * D_DIM + sseg * 16;      // halves within tile block
    const int    lstage = srow * LDS_PAD + sseg * 16;            // halves within LDS buffer

    // Async stage of one 16-embed tile (hi+lo) into LDS buffer `buf`.
    // GLOBAL_LOAD_ASYNC_TO_LDS_B128: IOFFSET adds to both global and LDS addr.
    auto stage = [&](int t, int buf) {
        unsigned long long ah = (unsigned long long)(const void*)(eh + (size_t)t * 16 * D_DIM + gstage);
        unsigned long long al = (unsigned long long)(const void*)(el + (size_t)t * 16 * D_DIM + gstage);
        unsigned dh = (unsigned)(unsigned long long)(const void*)&ehT[buf * TILE_H + lstage];
        unsigned dl = (unsigned)(unsigned long long)(const void*)&elT[buf * TILE_H + lstage];
        asm volatile(
            "global_load_async_to_lds_b128 %0, %2, off\n\t"
            "global_load_async_to_lds_b128 %0, %2, off offset:16\n\t"
            "global_load_async_to_lds_b128 %1, %3, off\n\t"
            "global_load_async_to_lds_b128 %1, %3, off offset:16"
            :: "v"(dh), "v"(dl), "v"(ah), "v"(al)
            : "memory");
    };

    // ---- Build B fragments: this lane's x row, 8 K-slices of 32, f16 hi/lo.
    // B map: lane-half selects contiguous K block of 16; elements sequential.
    stage(0, 0);   // prologue: kick off tile 0 while we convert x
    v16h bxh[8], bxl[8];
    #pragma unroll
    for (int s = 0; s < 8; ++s) {
        const float* xp = x + (size_t)xrow * D_DIM + s * 32 + half * 16;
        v4f f0 = *(const v4f*)(xp + 0);
        v4f f1 = *(const v4f*)(xp + 4);
        v4f f2 = *(const v4f*)(xp + 8);
        v4f f3 = *(const v4f*)(xp + 12);
        #pragma unroll
        for (int e2 = 0; e2 < 16; ++e2) {
            float fv = (e2 < 4) ? f0[e2] : (e2 < 8) ? f1[e2 - 4]
                     : (e2 < 12) ? f2[e2 - 8] : f3[e2 - 12];
            _Float16 h = (_Float16)fv;
            bxh[s][e2] = h;
            bxl[s][e2] = (_Float16)(fv - (float)h);
        }
    }

    // ---- Per-lane running top-8 (sorted ascending) in registers.
    float td[TOPK];
    int   ti[TOPK];
    #pragma unroll
    for (int k = 0; k < TOPK; ++k) { td[k] = 3.0e38f; ti[k] = 0; }

    const int numTiles = M >> 4;   // 16 embeds per tile
    for (int t = 0; t < numTiles; ++t) {
        // All waves done computing tile t-1 (same buffer we stage t+1 into).
        __syncthreads();
        if (t + 1 < numTiles) {
            stage(t + 1, (t + 1) & 1);
            asm volatile("s_wait_asynccnt 0x4" ::: "memory");  // retire stage(t)
        } else {
            asm volatile("s_wait_asynccnt 0x0" ::: "memory");
        }
        __syncthreads();           // stage(t) visible to all waves

        const _Float16* bh = &ehT[(t & 1) * TILE_H];
        const _Float16* bl = &elT[(t & 1) * TILE_H];

        // ---- 16x16 dot tile: 8 K-slices x 3 WMMAs (hh, hl, lh).
        v8f acc = {0.f, 0.f, 0.f, 0.f, 0.f, 0.f, 0.f, 0.f};
        #pragma unroll
        for (int s = 0; s < 8; ++s) {
            // A map (contiguous, matches B): elems 0..15 -> K = s*32+half*16+{0..15}.
            const _Float16* ap = &bh[lrow * LDS_PAD + s * 32 + half * 16];
            const _Float16* lp = &bl[lrow * LDS_PAD + s * 32 + half * 16];
            v8h a0 = *(const v8h*)(ap);
            v8h a1 = *(const v8h*)(ap + 8);
            v8h l0 = *(const v8h*)(lp);
            v8h l1 = *(const v8h*)(lp + 8);
            v16h A, L;
            #pragma unroll
            for (int e2 = 0; e2 < 8; ++e2) {
                A[e2] = a0[e2]; A[e2 + 8] = a1[e2];
                L[e2] = l0[e2]; L[e2 + 8] = l1[e2];
            }
            acc = __builtin_amdgcn_wmma_f32_16x16x32_f16(false, A, false, bxh[s],
                                                         (short)0, acc, false, false);
            acc = __builtin_amdgcn_wmma_f32_16x16x32_f16(false, A, false, bxl[s],
                                                         (short)0, acc, false, false);
            acc = __builtin_amdgcn_wmma_f32_16x16x32_f16(false, L, false, bxh[s],
                                                         (short)0, acc, false, false);
        }

        // ---- Candidates: dist = ||e||^2 - 2 x.e  (||x||^2 constant per row).
        const v4f* enp = (const v4f*)(enorm + t * 16 + half * 8);
        v4f en0 = enp[0], en1 = enp[1];
        #pragma unroll
        for (int v = 0; v < 8; ++v) {
            float dist = ((v < 4) ? en0[v] : en1[v - 4]) - 2.0f * acc[v];
            int ci = t * 16 + half * 8 + v;           // global embed index
            if (dist < td[TOPK - 1]) {                // rare after warm-up
                bool b[TOPK];
                #pragma unroll
                for (int j = 0; j < TOPK; ++j) b[j] = dist < td[j];
                float nd[TOPK]; int ni[TOPK];
                nd[0] = fminf(td[0], dist);
                ni[0] = b[0] ? ci : ti[0];
                #pragma unroll
                for (int j = 1; j < TOPK; ++j) {
                    nd[j] = fminf(td[j], fmaxf(td[j - 1], dist));
                    ni[j] = b[j] ? (b[j - 1] ? ti[j - 1] : ci) : ti[j];
                }
                #pragma unroll
                for (int j = 0; j < TOPK; ++j) { td[j] = nd[j]; ti[j] = ni[j]; }
            }
        }
    }

    // ---- Merge the two lanes (lane, lane+16) holding the same x-row.
    #pragma unroll
    for (int k = 0; k < TOPK; ++k) { md[wave][lane][k] = td[k]; mi[wave][lane][k] = ti[k]; }
    __syncthreads();
    if (lane < 16) {
        int ia = 0, ib = 0;
        #pragma unroll
        for (int k = 0; k < TOPK; ++k) {
            float da = md[wave][lane][ia];
            float db = md[wave][lane + 16][ib];
            int   ea = mi[wave][lane][ia];
            int   eb = mi[wave][lane + 16][ib];
            bool takeA = (da < db) || (da == db && ea < eb);
            int sel = takeA ? ea : eb;
            ridx[wave][lane][k] = sel;
            outi[(size_t)(rowbase + lane) * TOPK + k] = sel;
            ia += takeA ? 1 : 0;
            ib += takeA ? 0 : 1;
        }
    }
    __syncthreads();

    // ---- group_emb[d] = sum_k e_k[d]^2 / sum_k e_k[d]  (fp32, L2-resident embed)
    for (int r = 0; r < 16; ++r) {
        int ids[TOPK];
        #pragma unroll
        for (int k = 0; k < TOPK; ++k) ids[k] = ridx[wave][r][k];
        #pragma unroll
        for (int c = 0; c < 8; ++c) {
            int d = c * 32 + lane;
            float s1 = 0.f, s2 = 0.f;
            #pragma unroll
            for (int k = 0; k < TOPK; ++k) {
                float ev = embed[(size_t)ids[k] * D_DIM + d];
                s1 += ev;
                s2 += ev * ev;
            }
            outg[(size_t)(rowbase + r) * D_DIM + d] = s2 / s1;
        }
    }
}

// ---------------------------------------------------------------------------
extern "C" void kernel_launch(void* const* d_in, const int* in_sizes, int n_in,
                              void* d_out, int out_size, void* d_ws, size_t ws_size,
                              hipStream_t stream)
{
    const float* x     = (const float*)d_in[0];   // [N,256]
    const float* embed = (const float*)d_in[1];   // [M,256]
    const int N = in_sizes[0] / D_DIM;            // 65536
    const int M = in_sizes[1] / D_DIM;            // 2048

    // Workspace layout: enorm[M] f32 | eh[M*256] f16 | el[M*256] f16  (~2.01 MB)
    float*    enorm = (float*)d_ws;
    _Float16* eh    = (_Float16*)((char*)d_ws + (size_t)M * sizeof(float));
    _Float16* el    = eh + (size_t)M * D_DIM;

    float* outg = (float*)d_out;
    int*   outi = (int*)(outg + (size_t)N * D_DIM);

    prep_embed<<<M, 256, 0, stream>>>(embed, enorm, eh, el);
    interest_main<<<N / 128, 256, 0, stream>>>(x, embed, enorm, eh, el, outg, outi, M);
}
